// FusedGromovWasserstein_53257594470652
// MI455X (gfx1250) — compile-verified, matrix operand order checked by reading, not run on hardware
//
#include <hip/hip_runtime.h>
#include <hip/hip_bf16.h>

typedef __attribute__((ext_vector_type(16))) __bf16 v16bf;
typedef __attribute__((ext_vector_type(8)))  __bf16 v8bf;
typedef __attribute__((ext_vector_type(8)))  float  v8f;
typedef __attribute__((ext_vector_type(2)))  float  v2f;

#define D_DIM 256
#define LDSW  264   // padded bf16 row stride (multiple of 8 -> 16B aligned chunks)

#define SHUF16(x, y) __builtin_shufflevector((x), (y), 0,1,2,3,4,5,6,7,8,9,10,11,12,13,14,15)

// ---------------------------------------------------------------------------
// Kernel 0: convert the two 256x256 weight matrices to bf16 in workspace.
// ---------------------------------------------------------------------------
__global__ __launch_bounds__(256)
void cvt_weights_kernel(const float* __restrict__ W1, const float* __restrict__ W2,
                        __bf16* __restrict__ W1b, __bf16* __restrict__ W2b) {
  int i = blockIdx.x * blockDim.x + threadIdx.x;
  if (i < D_DIM * D_DIM) {
    W1b[i] = (__bf16)W1[i];
    W2b[i] = (__bf16)W2[i];
  }
}

// ---------------------------------------------------------------------------
// Kernel 1: per-batch fused MLP projection (2 layers, ReLU) for q and r,
// then C = cdist(pq, pr) via WMMA cross-dot + f32 norms.
// Block = 256 threads (8 waves), one block per batch.
// ---------------------------------------------------------------------------
__global__ __launch_bounds__(256)
void proj_cdist_kernel(const float* __restrict__ Xq_g, const float* __restrict__ Xr_g,
                       const __bf16* __restrict__ W1b, const float* __restrict__ b1,
                       const __bf16* __restrict__ W2b, const float* __restrict__ b2,
                       float* __restrict__ Cout) {
  __shared__ __attribute__((aligned(16))) __bf16 Xs[2][16][LDSW]; // X, later P
  __shared__ __attribute__((aligned(16))) __bf16 Hs[2][16][LDSW]; // hidden
  __shared__ float norms[2][16];

  const int b    = blockIdx.x;
  const int t    = threadIdx.x;
  const int lane = t & 31;
  const int wave = t >> 5;
  const int lo   = lane & 15;
  const int hi   = lane >> 4;

  // ---- stage inputs (f32 -> bf16) into LDS: 16 rows x 256 per input ----
  {
    const int row   = t >> 4;        // 0..15
    const int dbase = (t & 15) * 16; // 16 consecutive d per thread
    const float* sq = Xq_g + (size_t)b * 4096 + row * 256 + dbase;
    const float* sr = Xr_g + (size_t)b * 4096 + row * 256 + dbase;
    #pragma unroll
    for (int j = 0; j < 16; j += 4) {
      float4 vq = *(const float4*)(sq + j);
      float4 vr = *(const float4*)(sr + j);
      Xs[0][row][dbase + j + 0] = (__bf16)vq.x; Xs[0][row][dbase + j + 1] = (__bf16)vq.y;
      Xs[0][row][dbase + j + 2] = (__bf16)vq.z; Xs[0][row][dbase + j + 3] = (__bf16)vq.w;
      Xs[1][row][dbase + j + 0] = (__bf16)vr.x; Xs[1][row][dbase + j + 1] = (__bf16)vr.y;
      Xs[1][row][dbase + j + 2] = (__bf16)vr.z; Xs[1][row][dbase + j + 3] = (__bf16)vr.w;
    }
  }
  __syncthreads();

  // ---- layer 1: H = relu(X @ W1^T + b1), 32 tiles of 16 cols, 4 per wave ----
  for (int j = 0; j < 4; ++j) {
    const int tile = wave * 4 + j;          // 0..31
    const int inp  = tile >> 4;             // 0=q, 1=r
    const int ng   = (tile & 15) * 16 + lo; // output column (e)
    v8f acc = {};
    __builtin_prefetch(W2b + ng * 256, 0, 1); // warm L2 for layer 2 weights
    #pragma unroll
    for (int s = 0; s < 8; ++s) {
      const int kb = s * 32 + hi * 8;
      v8bf alo = *(const v8bf*)&Xs[inp][lo][kb];
      v8bf ahi = *(const v8bf*)&Xs[inp][lo][kb + 16];
      v8bf blo = *(const v8bf*)(W1b + ng * 256 + kb);
      v8bf bhi = *(const v8bf*)(W1b + ng * 256 + kb + 16);
      v16bf av = SHUF16(alo, ahi);
      v16bf bv = SHUF16(blo, bhi);
      acc = __builtin_amdgcn_wmma_f32_16x16x32_bf16(false, av, false, bv, (short)0, acc, false, false);
    }
    const float bias = b1[ng];
    #pragma unroll
    for (int g = 0; g < 8; ++g) {
      float v = acc[g] + bias;
      v = v > 0.f ? v : 0.f;
      Hs[inp][g + 8 * hi][ng] = (__bf16)v;
    }
  }
  __syncthreads();

  // ---- layer 2: P = H @ W2^T + b2, written back into Xs ----
  for (int j = 0; j < 4; ++j) {
    const int tile = wave * 4 + j;
    const int inp  = tile >> 4;
    const int ng   = (tile & 15) * 16 + lo;
    v8f acc = {};
    #pragma unroll
    for (int s = 0; s < 8; ++s) {
      const int kb = s * 32 + hi * 8;
      v8bf alo = *(const v8bf*)&Hs[inp][lo][kb];
      v8bf ahi = *(const v8bf*)&Hs[inp][lo][kb + 16];
      v8bf blo = *(const v8bf*)(W2b + ng * 256 + kb);
      v8bf bhi = *(const v8bf*)(W2b + ng * 256 + kb + 16);
      v16bf av = SHUF16(alo, ahi);
      v16bf bv = SHUF16(blo, bhi);
      acc = __builtin_amdgcn_wmma_f32_16x16x32_bf16(false, av, false, bv, (short)0, acc, false, false);
    }
    const float bias = b2[ng];
    #pragma unroll
    for (int g = 0; g < 8; ++g)
      Xs[inp][g + 8 * hi][ng] = (__bf16)(acc[g] + bias);
  }
  __syncthreads();

  // ---- row norms of pq, pr ----
  if (t < 32) {
    const int inp = t >> 4, row = t & 15;
    float s = 0.f;
    for (int d = 0; d < D_DIM; ++d) { float v = (float)Xs[inp][row][d]; s += v * v; }
    norms[inp][row] = s;
  }
  __syncthreads();

  // ---- C[k,m] = sqrt(max(|pq_k|^2 + |pr_m|^2 - 2 pq_k.pr_m, 1e-6)) ----
  if (wave == 0) {
    v8f acc = {};
    #pragma unroll
    for (int s = 0; s < 8; ++s) {
      const int kb = s * 32 + hi * 8;
      v8bf alo = *(const v8bf*)&Xs[0][lo][kb];
      v8bf ahi = *(const v8bf*)&Xs[0][lo][kb + 16];
      v8bf blo = *(const v8bf*)&Xs[1][lo][kb];  // B[k][m] = pr[m][k], m = lo
      v8bf bhi = *(const v8bf*)&Xs[1][lo][kb + 16];
      v16bf av = SHUF16(alo, ahi);
      v16bf bv = SHUF16(blo, bhi);
      acc = __builtin_amdgcn_wmma_f32_16x16x32_bf16(false, av, false, bv, (short)0, acc, false, false);
    }
    #pragma unroll
    for (int g = 0; g < 8; ++g) {
      const int row = g + 8 * hi;
      float val = sqrtf(fmaxf(norms[0][row] + norms[1][lo] - 2.f * acc[g], 1e-6f));
      Cout[(size_t)b * 256 + row * 16 + lo] = val;
    }
  }
}

// ---------------------------------------------------------------------------
// Kernel 2: per-batch FGW outer loop + Sinkhorn. One wave per batch
// (8 waves / block, wave-private LDS slices). Sq@T@Sr via f32 WMMA 16x16x4.
// ---------------------------------------------------------------------------
#define NW 8
__global__ __launch_bounds__(256)
void fgw_kernel(const float* __restrict__ Cin,
                const float* __restrict__ mask_q, const float* __restrict__ mask_r,
                const float* __restrict__ cq, const float* __restrict__ cr,
                const float* __restrict__ log_eps_p,
                float* __restrict__ score, float* __restrict__ Tout,
                float* __restrict__ costOut) {
  __shared__ float Sq_t[NW][16][17];
  __shared__ float Sr_t[NW][16][17];
  __shared__ float Tt  [NW][16][17];
  __shared__ float Sc  [NW][16][17];
  __shared__ float rsum[NW][16], csum[NW][16];
  __shared__ float t1a [NW][16], t2a [NW][16];
  __shared__ float lua [NW][16], lva [NW][16];
  __shared__ float red [NW][32];

  const int t    = threadIdx.x;
  const int w    = t >> 5;
  const int lane = t & 31;
  const int lo   = lane & 15;
  const int hi   = lane >> 4;
  const int b    = blockIdx.x * NW + w;

  const float eps = __expf(log_eps_p[0]);
  const float rho = 0.1f / (0.1f + eps);

  float sumq = 0.f, sumr = 0.f;
  for (int i = 0; i < 16; ++i) { sumq += mask_q[b * 16 + i]; sumr += mask_r[b * 16 + i]; }
  const float inv_q = 1.f / (sumq + 1e-8f);
  const float inv_r = 1.f / (sumr + 1e-8f);

  // ---- structure matrices Sq, Sr from 2D centroids ----
  {
    const float bqx = cq[(size_t)b * 32 + lo * 2], bqy = cq[(size_t)b * 32 + lo * 2 + 1];
    const float brx = cr[(size_t)b * 32 + lo * 2], bry = cr[(size_t)b * 32 + lo * 2 + 1];
    #pragma unroll
    for (int g = 0; g < 8; ++g) {
      const int row = g + 8 * hi;
      float ax = cq[(size_t)b * 32 + row * 2], ay = cq[(size_t)b * 32 + row * 2 + 1];
      float dx = ax - bqx, dy = ay - bqy;
      Sq_t[w][row][lo] = sqrtf(fmaxf(dx * dx + dy * dy, 1e-6f));
      ax = cr[(size_t)b * 32 + row * 2]; ay = cr[(size_t)b * 32 + row * 2 + 1];
      dx = ax - brx; dy = ay - bry;
      Sr_t[w][row][lo] = sqrtf(fmaxf(dx * dx + dy * dy, 1e-6f));
    }
  }

  float cfeat[8], Tf[8], cfgw[8], logK[8], lu[8];
  const float nuv = mask_r[b * 16 + lo] * inv_r;
  #pragma unroll
  for (int g = 0; g < 8; ++g) {
    const int row = g + 8 * hi;
    cfeat[g] = Cin[(size_t)b * 256 + row * 16 + lo];
    Tf[g]    = (mask_q[b * 16 + row] * inv_q) * nuv;
    cfgw[g]  = 0.f;
  }
  __syncthreads();

  for (int it = 0; it < 5; ++it) {
    #pragma unroll
    for (int g = 0; g < 8; ++g) Tt[w][g + 8 * hi][lo] = Tf[g];
    __syncthreads();
    if (hi == 0) { float s = 0.f; for (int c = 0; c < 16; ++c) s += Tt[w][lo][c]; rsum[w][lo] = s; }
    else         { float s = 0.f; for (int r = 0; r < 16; ++r) s += Tt[w][r][lo]; csum[w][lo] = s; }
    __syncthreads();
    if (hi == 0) { float s = 0.f; for (int l = 0; l < 16; ++l) { float q = Sq_t[w][lo][l]; s += q * q * rsum[w][l]; } t1a[w][lo] = s; }
    else         { float s = 0.f; for (int m = 0; m < 16; ++m) { float q = Sr_t[w][m][lo]; s += csum[w][m] * q * q; } t2a[w][lo] = s; }
    __syncthreads();

    // P1 = T @ Sr  (four chained f32 16x16x4 WMMAs)
    v8f p = {};
    #pragma unroll
    for (int s4 = 0; s4 < 4; ++s4) {
      v2f av = {}, bv = {};
      #pragma unroll
      for (int g = 0; g < 2; ++g) {
        const int k = s4 * 4 + g + 2 * hi;
        av[g] = Tt[w][lo][k];
        bv[g] = Sr_t[w][k][lo];
      }
      p = __builtin_amdgcn_wmma_f32_16x16x4_f32(false, av, false, bv, (short)0, p, false, false);
    }
    #pragma unroll
    for (int g = 0; g < 8; ++g) Sc[w][g + 8 * hi][lo] = p[g];
    __syncthreads();

    // P2 = Sq @ P1
    v8f q2 = {};
    #pragma unroll
    for (int s4 = 0; s4 < 4; ++s4) {
      v2f av = {}, bv = {};
      #pragma unroll
      for (int g = 0; g < 2; ++g) {
        const int k = s4 * 4 + g + 2 * hi;
        av[g] = Sq_t[w][lo][k];
        bv[g] = Sc[w][k][lo];
      }
      q2 = __builtin_amdgcn_wmma_f32_16x16x4_f32(false, av, false, bv, (short)0, q2, false, false);
    }

    #pragma unroll
    for (int g = 0; g < 8; ++g) {
      const int row = g + 8 * hi;
      const float lgw = t1a[w][row] + t2a[w][lo] - 2.f * q2[g];
      cfgw[g] = 0.5f * cfeat[g] + 0.5f * lgw;   // LAMBDA_FGW = 0.5
      logK[g] = -cfgw[g] / eps;
    }

    // ---- Sinkhorn (10 iterations) ----
    float lv = 0.f;
    #pragma unroll
    for (int g = 0; g < 8; ++g) lu[g] = 0.f;
    for (int si = 0; si < 10; ++si) {
      #pragma unroll
      for (int g = 0; g < 8; ++g) Tt[w][g + 8 * hi][lo] = logK[g] + lv;
      __syncthreads();
      if (hi == 0) {  // row LSE over columns
        float m = -1e30f;
        for (int c = 0; c < 16; ++c) m = fmaxf(m, Tt[w][lo][c]);
        float s = 0.f;
        for (int c = 0; c < 16; ++c) s += __expf(Tt[w][lo][c] - m);
        const float lse = m + __logf(s);
        const float lmu = __logf(fmaxf(mask_q[b * 16 + lo] * inv_q, 1e-8f));
        lua[w][lo] = rho * (lmu - lse);
      }
      __syncthreads();
      #pragma unroll
      for (int g = 0; g < 8; ++g) lu[g] = lua[w][g + 8 * hi];
      #pragma unroll
      for (int g = 0; g < 8; ++g) Tt[w][g + 8 * hi][lo] = logK[g] + lu[g];
      __syncthreads();
      if (hi == 0) {  // column LSE over rows
        float m = -1e30f;
        for (int r = 0; r < 16; ++r) m = fmaxf(m, Tt[w][r][lo]);
        float s = 0.f;
        for (int r = 0; r < 16; ++r) s += __expf(Tt[w][r][lo] - m);
        const float lse = m + __logf(s);
        const float lnu = __logf(fmaxf(mask_r[b * 16 + lo] * inv_r, 1e-8f));
        lva[w][lo] = rho * (lnu - lse);
      }
      __syncthreads();
      lv = lva[w][lo];
    }
    #pragma unroll
    for (int g = 0; g < 8; ++g) Tf[g] = __expf(lu[g] + logK[g] + lv);
    __syncthreads();
  }

  // ---- cost = sum(T * C_fgw), outputs ----
  float part = 0.f;
  #pragma unroll
  for (int g = 0; g < 8; ++g) part += Tf[g] * cfgw[g];
  red[w][lane] = part;
  __syncthreads();
  if (lane == 0) {
    float tot = 0.f;
    for (int i = 0; i < 32; ++i) tot += red[w][i];
    costOut[b] = tot;
    score[b]   = 1.f / (1.f + __expf(tot));   // sigmoid(-cost)
  }
  #pragma unroll
  for (int g = 0; g < 8; ++g)
    Tout[(size_t)b * 256 + (g + 8 * hi) * 16 + lo] = Tf[g];
}

// ---------------------------------------------------------------------------
extern "C" void kernel_launch(void* const* d_in, const int* in_sizes, int n_in,
                              void* d_out, int out_size, void* d_ws, size_t ws_size,
                              hipStream_t stream) {
  const float* slots_q = (const float*)d_in[0];
  const float* slots_r = (const float*)d_in[1];
  const float* mask_q  = (const float*)d_in[2];
  const float* mask_r  = (const float*)d_in[3];
  const float* cent_q  = (const float*)d_in[4];
  const float* cent_r  = (const float*)d_in[5];
  const float* W1      = (const float*)d_in[6];
  const float* b1      = (const float*)d_in[7];
  const float* W2      = (const float*)d_in[8];
  const float* b2      = (const float*)d_in[9];
  const float* log_eps = (const float*)d_in[10];

  const int B = in_sizes[0] / (16 * 256);

  __bf16* W1b = (__bf16*)d_ws;
  __bf16* W2b = W1b + D_DIM * D_DIM;

  float* out   = (float*)d_out;
  float* score = out;                        // [B]
  float* Tout  = out + B;                    // [B,16,16]
  float* Cout  = Tout + (size_t)B * 256;     // [B,16,16]
  float* costO = Cout + (size_t)B * 256;     // [B]

  cvt_weights_kernel<<<(D_DIM * D_DIM + 255) / 256, 256, 0, stream>>>(W1, W2, W1b, W2b);
  proj_cdist_kernel<<<B, 256, 0, stream>>>(slots_q, slots_r, W1b, b1, W2b, b2, Cout);
  fgw_kernel<<<B / NW, 256, 0, stream>>>(Cout, mask_q, mask_r, cent_q, cent_r, log_eps,
                                         score, Tout, costO);
}
